// DeltaNet_31877247271426
// MI455X (gfx1250) — compile-verified, matrix operand order checked by reading, not run on hardware
//
#include <hip/hip_runtime.h>
#include <stdint.h>

#define DEVINL __device__ __forceinline__

typedef __bf16 v16bf __attribute__((ext_vector_type(16)));
typedef float  v8f   __attribute__((ext_vector_type(8)));
typedef int    v4i   __attribute__((vector_size(16)));

constexpr int B_  = 4, H_ = 8, L_ = 4096, DK = 128, DV = 128, C = 32;
constexpr int NCH  = L_ / C;       // 128 chunks per head-seq
constexpr int BH   = B_ * H_;      // 32
constexpr int NTOT = BH * NCH;     // 4096 chunk tiles

DEVINL v8f wmma_bf16(v16bf a, v16bf b, v8f c) {
  return __builtin_amdgcn_wmma_f32_16x16x32_bf16(false, a, false, b, (short)0, c,
                                                 false, false);
}

// ---------------- async global -> LDS staging (CDNA5 ASYNCcnt path) ---------
#if defined(__AMDGCN__) && __has_builtin(__builtin_amdgcn_global_load_async_to_lds_b128)
#define HAVE_ASYNC_LDS 1
#else
#define HAVE_ASYNC_LDS 0
#endif

DEVINL void cp16_to_lds(void* l, const void* g) {
#if HAVE_ASYNC_LDS
  __builtin_amdgcn_global_load_async_to_lds_b128((v4i*)g, (v4i*)l, 0, 0);
#else
  *(int4*)l = *(const int4*)g;
#endif
}
DEVINL void async_stage_fence() {
#if HAVE_ASYNC_LDS
  asm volatile("s_wait_asynccnt 0" ::: "memory");
#endif
}

// ---- fragment gathers (wave32, ISA 7.12.2 layouts) ----
// A: 16x32 tile of row-major MxK; lane&15 = row, lane>>4 selects K-halves.
// Two contiguous 16B runs per lane -> ds_load_b128.
DEVINL v16bf ldfrag_a(const __bf16* base, int ld, int row0, int k0, int lane) {
  v16bf f;
  const int m  = row0 + (lane & 15);
  const int kb = k0 + ((lane >> 4) << 3);
  const __bf16* p = base + m * ld;
#pragma unroll
  for (int e = 0; e < 8; ++e) f[e] = p[kb + e];
#pragma unroll
  for (int e = 0; e < 8; ++e) f[8 + e] = p[16 + kb + e];
  return f;
}
// B from [N][K]-layout storage: logical B[kk][n] = base[n*ld + kk].
// Also two contiguous 16B runs per lane.
DEVINL v16bf ldfrag_b_t(const __bf16* base, int ld, int k0, int col0, int lane) {
  v16bf f;
  const int n  = col0 + (lane & 15);
  const int kb = k0 + ((lane >> 4) << 3);
  const __bf16* p = base + n * ld;
#pragma unroll
  for (int e = 0; e < 8; ++e) f[e] = p[kb + e];
#pragma unroll
  for (int e = 0; e < 8; ++e) f[8 + e] = p[16 + kb + e];
  return f;
}
// C/D 16x16 f32: VGPR r -> M = r + 8*(lane>=16), N = lane&15
DEVINL void stfrag_f32(float* base, int ld, int row0, int col0, int lane, v8f d) {
  const int n  = col0 + (lane & 15);
  const int mb = row0 + ((lane >> 4) << 3);
#pragma unroll
  for (int r = 0; r < 8; ++r) base[(mb + r) * ld + n] = d[r];
}

// =====================================================================
// Pass 1: per-chunk preprocess (fully parallel over 4096 chunk tiles).
// All operands destined to be WMMA B-operands (or transposed A-operands)
// are built in [N][K] orientation so every fragment gather is ds_load_b128.
// =====================================================================
__global__ __launch_bounds__(256)
void dn_preprocess(const float* __restrict__ q, const float* __restrict__ k,
                   const float* __restrict__ v, const float* __restrict__ beta,
                   __bf16* __restrict__ ws_q, __bf16* __restrict__ ws_kT,
                   __bf16* __restrict__ ws_w, float* __restrict__ ws_u,
                   __bf16* __restrict__ ws_a) {
  __shared__ __align__(16) __bf16 qb[C][DK];       // A-use (attn)
  __shared__ __align__(16) __bf16 kb[C][DK];       // [n][k] for B=k^T uses
  __shared__ __align__(16) __bf16 kbT[DK][C];      // k transposed: export for scan
  __shared__ __align__(16) __bf16 kbeta[C][DK];    // A-use
  __shared__ __align__(16) __bf16 kbetaT[DK][C];   // [n][k] B-use for w = T@k_b
  __shared__ __align__(16) __bf16 vbT[DV][C];      // [n][k] B-use for u = T@v_b
  __shared__ float  Amat[C][C + 1];
  __shared__ __align__(16) __bf16 Tb[C][C];
  __shared__ float  psum[C][8];
  __shared__ float  rn[C];
  __shared__ float  betas[C];

  const int tid  = threadIdx.x;
  const int lane = tid & 31;
  const int wv   = tid >> 5;
  const int g    = blockIdx.x;
  const int bh   = g / NCH;
  const int ch   = g % NCH;
  const size_t base = ((size_t)bh * L_ + (size_t)ch * C) * DK;
  const int r  = tid >> 3;   // row 0..31
  const int sg = tid & 7;    // 16-wide segment 0..7
  const int e0 = sg * 16;

  if (tid < C) betas[tid] = beta[(size_t)bh * L_ + (size_t)ch * C + tid];

  // ---- q: load + rownorm ----
  float qv[16];
  {
    const float* p = q + base + (size_t)r * DK + e0;
    float s = 0.f;
#pragma unroll
    for (int e = 0; e < 16; ++e) { qv[e] = p[e]; s += qv[e] * qv[e]; }
    psum[r][sg] = s;
  }
  __syncthreads();
  if (tid < C) {
    float s = 0.f;
#pragma unroll
    for (int i = 0; i < 8; ++i) s += psum[tid][i];
    rn[tid] = rsqrtf(s + 1e-6f);
  }
  __syncthreads();
  // ---- write qb; k: load + rownorm ----
  float kv[16];
  {
    const float rr = rn[r];
#pragma unroll
    for (int e = 0; e < 16; ++e) qb[r][e0 + e] = (__bf16)(qv[e] * rr);
    const float* p = k + base + (size_t)r * DK + e0;
    float s = 0.f;
#pragma unroll
    for (int e = 0; e < 16; ++e) { kv[e] = p[e]; s += kv[e] * kv[e]; }
    psum[r][sg] = s;
  }
  __syncthreads();
  if (tid < C) {
    float s = 0.f;
#pragma unroll
    for (int i = 0; i < 8; ++i) s += psum[tid][i];
    rn[tid] = rsqrtf(s + 1e-6f);
  }
  __syncthreads();
  {
    const float rr = rn[r];
    const float bt = betas[r];
#pragma unroll
    for (int e = 0; e < 16; ++e) {
      const float kn = kv[e] * rr;
      const float kbv = kn * bt;
      kb[r][e0 + e]      = (__bf16)kn;
      kbT[e0 + e][r]     = (__bf16)kn;
      kbeta[r][e0 + e]   = (__bf16)kbv;
      kbetaT[e0 + e][r]  = (__bf16)kbv;
    }
    const float* p = v + base + (size_t)r * DK + e0;
#pragma unroll
    for (int e = 0; e < 16; ++e) vbT[e0 + e][r] = (__bf16)(p[e] * bt);
  }
  __syncthreads();

  // ---- waves 0-3: A = -(k_beta @ k^T) strict-lower; waves 4-7: attn ----
  if (wv < 4) {
    const int mi = (wv >> 1) * 16, ni = (wv & 1) * 16;
    v8f acc = {};
#pragma unroll
    for (int kk = 0; kk < DK; kk += 32) {
      v16bf a = ldfrag_a(&kbeta[0][0], DK, mi, kk, lane);
      v16bf b = ldfrag_b_t(&kb[0][0], DK, kk, ni, lane);   // B = k^T
      acc = wmma_bf16(a, b, acc);
    }
    const int n  = ni + (lane & 15);
    const int mb = mi + ((lane >> 4) << 3);
#pragma unroll
    for (int r2 = 0; r2 < 8; ++r2) {
      const int m = mb + r2;
      Amat[m][n] = (m > n) ? -acc[r2] : 0.f;
    }
  } else {
    const int wz = wv - 4;
    const int mi = (wz >> 1) * 16, ni = (wz & 1) * 16;
    v8f acc = {};
#pragma unroll
    for (int kk = 0; kk < DK; kk += 32) {
      v16bf a = ldfrag_a(&qb[0][0], DK, mi, kk, lane);
      v16bf b = ldfrag_b_t(&kb[0][0], DK, kk, ni, lane);
      acc = wmma_bf16(a, b, acc);
    }
    __bf16* ap   = ws_a + (size_t)g * C * C;
    const int n  = ni + (lane & 15);
    const int mb = mi + ((lane >> 4) << 3);
#pragma unroll
    for (int r2 = 0; r2 < 8; ++r2) {
      const int m = mb + r2;
      ap[m * C + n] = (__bf16)((m >= n) ? acc[r2] : 0.f);  // lower-inclusive mask
    }
  }
  __syncthreads();

  // ---- forward substitution (sequential, wave 0 in LDS) ----
  for (int i = 1; i < C; ++i) {
    if (wv == 0) {
      float s = 0.f;
      for (int kk = 0; kk < C; ++kk) s += Amat[i][kk] * Amat[kk][lane];
      if (lane < i) Amat[i][lane] += s;
    }
    __syncthreads();
  }
  for (int idx = tid; idx < C * C; idx += 256) {
    const int r2 = idx >> 5, cc = idx & 31;
    Tb[r2][cc] = (__bf16)(Amat[r2][cc] + (r2 == cc ? 1.f : 0.f));
  }
  __syncthreads();

  // ---- u = T @ v_b (f32 out), w = T @ k_beta (bf16 out): 16 tiles / 8 waves --
  {
    float*  up = ws_u + (size_t)g * C * DV;
    __bf16* wp = ws_w + (size_t)g * C * DK;
    v8f z = {};
#pragma unroll
    for (int t = 0; t < 2; ++t) {
      const int tt = wv * 2 + t;            // 0..15
      const int mi = (tt & 1) * 16;
      const int ni = (tt >> 1) * 16;
      v16bf a  = ldfrag_a(&Tb[0][0], C, mi, 0, lane);
      v16bf bu = ldfrag_b_t(&vbT[0][0], C, 0, ni, lane);
      v8f du = wmma_bf16(a, bu, z);
      stfrag_f32(up, DV, mi, ni, lane, du);
      v16bf bw = ldfrag_b_t(&kbetaT[0][0], C, 0, ni, lane);
      v8f dw = wmma_bf16(a, bw, z);
      const int n  = ni + (lane & 15);
      const int mb = mi + ((lane >> 4) << 3);
#pragma unroll
      for (int r2 = 0; r2 < 8; ++r2) wp[(mb + r2) * DK + n] = (__bf16)dw[r2];
    }
  }

  // ---- stage normalized q (row-major) and k^T ([dk][c]) to workspace ----
  {
    const uint32_t* s1 = (const uint32_t*)&qb[0][0];
    const uint32_t* s2 = (const uint32_t*)&kbT[0][0];
    uint32_t* d1 = (uint32_t*)(ws_q + (size_t)g * C * DK);
    uint32_t* d2 = (uint32_t*)(ws_kT + (size_t)g * C * DK);
    for (int i = tid; i < C * DK / 2; i += 256) { d1[i] = s1[i]; d2[i] = s2[i]; }
  }
}

// =====================================================================
// Pass 2: chunk scan. One WG per (b,h, dv-tile of 32) -> 128 WGs, 4 waves.
// S fp32 lives in registers (v8f sreg[4] per wave = its 32 dk-rows of S),
// accumulated exactly through the WMMA C/D operand. bf16 image of S is
// kept in LDS [n][k] orientation, double-buffered across chunks.
// =====================================================================
__global__ __launch_bounds__(128)
void dn_scan(const __bf16* __restrict__ ws_q, const __bf16* __restrict__ ws_kT,
             const __bf16* __restrict__ ws_w, const float* __restrict__ ws_u,
             const __bf16* __restrict__ ws_a, float* __restrict__ out) {
  __shared__ __align__(16) __bf16 SbT[2][32][DK];  // S^T bf16 [dv n][dk k], x2
  __shared__ __align__(16) __bf16 wsh[C][DK];      // A-use
  __shared__ __align__(16) __bf16 qsh[C][DK];      // A-use
  __shared__ __align__(16) __bf16 kTsh[DK][C];     // A-use (k^T)
  __shared__ __align__(16) float  ush[C][32];
  __shared__ __align__(16) __bf16 attnsh[C][C];    // A-use
  __shared__ __align__(16) __bf16 uibT[32][C];     // u_i^T [dv n][c k]

  const int tid  = threadIdx.x;
  const int lane = tid & 31;
  const int wv   = tid >> 5;          // 0..3
  const int bh   = blockIdx.x >> 2;
  const int vt   = blockIdx.x & 3;
  const int c0   = vt * 32;           // dv column base

  for (int i = tid; i < 32 * DK; i += 128) (&SbT[0][0][0])[i] = (__bf16)0.f;

  v8f sreg[4] = {{}, {}, {}, {}};     // S tiles: dk rows [wv*32, wv*32+32)
  const int mi = (wv >> 1) * 16;      // chunk-row tile for this wave
  const int ni = (wv & 1) * 16;       // dv-col tile for this wave
  __syncthreads();

  for (int j = 0; j < NCH; ++j) {
    const size_t g = (size_t)bh * NCH + j;
    const int cur = j & 1, nxt = cur ^ 1;

    // ---- stage chunk tiles (async DMA to LDS; ASYNCcnt) ----
    {
      const char* gw = (const char*)(ws_w + g * C * DK);
      const char* gq = (const char*)(ws_q + g * C * DK);
      const char* gk = (const char*)(ws_kT + g * C * DK);
      const char* ga = (const char*)(ws_a + g * C * C);
      char* lw = (char*)&wsh[0][0];
      char* lq = (char*)&qsh[0][0];
      char* lk = (char*)&kTsh[0][0];
      char* la = (char*)&attnsh[0][0];
      for (int i = tid; i < 512; i += 128) {           // 8 KB each
        cp16_to_lds(lw + i * 16, gw + i * 16);
        cp16_to_lds(lq + i * 16, gq + i * 16);
        cp16_to_lds(lk + i * 16, gk + i * 16);
      }
      for (int i = tid; i < 128; i += 128)             // 2 KB attn
        cp16_to_lds(la + i * 16, ga + i * 16);
      const char* gu = (const char*)(ws_u + g * C * DV + c0);
      char* lu = (char*)&ush[0][0];
      for (int i = tid; i < 256; i += 128) {           // 32 rows x 128 B
        const int rr = i >> 3, oo = i & 7;
        cp16_to_lds(lu + i * 16, gu + (size_t)rr * DV * 4 + oo * 16);
      }
      if (j + 1 < NCH) {                               // global_prefetch_b8
        __builtin_prefetch(ws_w + (g + 1) * C * DK, 0, 1);
        __builtin_prefetch(ws_q + (g + 1) * C * DK, 0, 1);
        __builtin_prefetch(ws_kT + (g + 1) * C * DK, 0, 1);
      }
    }
    async_stage_fence();
    __syncthreads();

    // ---- u_i = u0 - w @ S : one 16x16 tile per wave ----
    {
      v8f acc = {};
#pragma unroll
      for (int kk = 0; kk < DK; kk += 32) {
        v16bf a = ldfrag_a(&wsh[0][0], DK, mi, kk, lane);
        v16bf b = ldfrag_b_t(&SbT[cur][0][0], DK, kk, ni, lane);
        acc = wmma_bf16(a, b, acc);
      }
      const int n  = ni + (lane & 15);
      const int mb = mi + ((lane >> 4) << 3);
#pragma unroll
      for (int r2 = 0; r2 < 8; ++r2)
        uibT[n][mb + r2] = (__bf16)(ush[mb + r2][n] - acc[r2]);
    }
    __syncthreads();

    // ---- out = q @ S + attn @ u_i : one tile per wave ----
    {
      v8f acc = {};
#pragma unroll
      for (int kk = 0; kk < DK; kk += 32) {
        v16bf a = ldfrag_a(&qsh[0][0], DK, mi, kk, lane);
        v16bf b = ldfrag_b_t(&SbT[cur][0][0], DK, kk, ni, lane);
        acc = wmma_bf16(a, b, acc);
      }
      v16bf aa = ldfrag_a(&attnsh[0][0], C, mi, 0, lane);
      v16bf bu = ldfrag_b_t(&uibT[0][0], C, 0, ni, lane);
      acc = wmma_bf16(aa, bu, acc);
      const int n  = ni + (lane & 15);
      const int mb = mi + ((lane >> 4) << 3);
      float* op = out + ((size_t)bh * L_ + (size_t)j * C) * DV + c0;
#pragma unroll
      for (int r2 = 0; r2 < 8; ++r2) op[(size_t)(mb + r2) * DV + n] = acc[r2];
    }

    // ---- S += k^T @ u_i : 4 register tiles per wave, write bf16 image ----
    {
#pragma unroll
      for (int t = 0; t < 4; ++t) {
        const int tm = wv * 32 + (t >> 1) * 16;   // dk row tile
        const int tn = (t & 1) * 16;              // dv col tile
        v16bf a = ldfrag_a(&kTsh[0][0], C, tm, 0, lane);   // A[m=dk][kk=c]
        v16bf b = ldfrag_b_t(&uibT[0][0], C, 0, tn, lane);
        sreg[t] = wmma_bf16(a, b, sreg[t]);
        const int n  = tn + (lane & 15);
        const int mb = tm + ((lane >> 4) << 3);
#pragma unroll
        for (int r2 = 0; r2 < 8; ++r2)
          SbT[nxt][n][mb + r2] = (__bf16)sreg[t][r2];
      }
    }
    __syncthreads();
  }
}

// =====================================================================
extern "C" void kernel_launch(void* const* d_in, const int* in_sizes, int n_in,
                              void* d_out, int out_size, void* d_ws, size_t ws_size,
                              hipStream_t stream) {
  const float* q    = (const float*)d_in[0];
  const float* k    = (const float*)d_in[1];
  const float* v    = (const float*)d_in[2];
  const float* beta = (const float*)d_in[3];
  float* out = (float*)d_out;

  const size_t nqk = (size_t)NTOT * C * DK;   // bf16 elems per q/kT/w plane
  __bf16* ws_q  = (__bf16*)d_ws;
  __bf16* ws_kT = ws_q + nqk;
  __bf16* ws_w  = ws_kT + nqk;
  float*  ws_u  = (float*)(ws_w + nqk);
  __bf16* ws_a  = (__bf16*)(ws_u + (size_t)NTOT * C * DV);

  dn_preprocess<<<NTOT, 256, 0, stream>>>(q, k, v, beta, ws_q, ws_kT, ws_w, ws_u, ws_a);
  dn_scan<<<BH * 4, 128, 0, stream>>>(ws_q, ws_kT, ws_w, ws_u, ws_a, out);
}
